// DQNSolverCNN6_90263032693456
// MI455X (gfx1250) — compile-verified, matrix operand order checked by reading, not run on hardware
//
#include <hip/hip_runtime.h>

typedef _Float16 v16h __attribute__((ext_vector_type(16)));
typedef _Float16 v8h  __attribute__((ext_vector_type(8)));
typedef float    v8f  __attribute__((ext_vector_type(8)));

#define WAVES 8
#define SROW  132          // f32 state row stride (padded to dodge bank conflicts)
#define HALT_T 0.99f

// ---------------- LDS layout (dynamic shared) ----------------
// S    : [96][132] f32 state                      @ 0        (50688 B)
// SUM  : [96][132] f32 state_sum                  @ 50688    (50688 B)
// U    : [96][128] f16 (LN out / attn-out reuse)  @ 101376   (24576 B)
// QH   : union: QKV [96][384] f16 / H [96][512]   @ 125952   (98304 B)
// SCR  : [96][96] f32 scores (+conv scratch)      @ 224256   (36864 B)
// P16  : [96][96] f16 probs                       @ 261120   (18432 B)
// MISC : gvec[256], hvec[32], gmu[32], gvar[32], scal[4] @ 279552
#define OFF_S    0
#define OFF_SUM  50688
#define OFF_U    101376
#define OFF_QH   125952
#define OFF_SCR  224256
#define OFF_P    261120
#define OFF_MISC 279552
#define SMEM_BYTES 281088

struct ActParams {
  const float *x;
  const float *cc_wr, *cc_br, *cc_wc, *cc_bc, *cc_wb, *cc_bb;
  const float *red_w, *red_b, *gn_s, *gn_b;
  const float *ln1_s, *ln1_b, *bqkv, *bo, *ln2_s, *ln2_b, *b1, *b2;
  const float *hg_W1, *hg_b1, *hg_W2, *hg_b2, *fc_W, *fc_b;
  const _Float16 *wqkv, *wo, *w1, *w2;   // pre-swizzled f16 B-fragments in ws
  float *out;
  int nb;
};

// ---------------- WMMA helpers ----------------
__device__ inline v8f wmma16(v16h a, v16h b, v8f c) {
  // (neg_a, A, neg_b, B, c_mod, C, reuse_a, reuse_b)
  return __builtin_amdgcn_wmma_f32_16x16x32_f16(false, a, false, b, (short)0, c,
                                                false, false);
}

// A/B fragment from a row-major f16 LDS matrix (row = M or token, contiguous = K).
// Lane l: row = l&15, K base kb = (l>>4)*8; elements 0..7 = K kb..kb+7,
// elements 8..15 = K kb+16..kb+23  (per ISA 16-bit 16x32 layout).
__device__ inline v16h load_frag(const _Float16* base, int stride, int lane) {
  int r  = lane & 15;
  int kb = (lane >> 4) << 3;
  const _Float16* pp = base + r * stride + kb;
  union { v16h v; v8h h[2]; } u;
  u.h[0] = *(const v8h*)(pp);
  u.h[1] = *(const v8h*)(pp + 16);
  return u.v;
}

// B fragment where K runs along rows of stride `kstride` (for P x V).
__device__ inline v16h load_frag_kstrided(const _Float16* base, int kstride, int lane) {
  int c  = lane & 15;
  int kb = (lane >> 4) << 3;
  v16h v;
  #pragma unroll
  for (int e = 0; e < 16; e++) {
    int k = kb + (e < 8 ? e : e + 8);
    v[e] = base[k * kstride + c];
  }
  return v;
}

// ---------------- weight swizzle: f32 [K][N] -> f16 WMMA-B tiles ----------------
__global__ __launch_bounds__(256) void swz_kernel(const float* __restrict__ W,
                                                  _Float16* __restrict__ out,
                                                  int K, int N) {
  int NT = N >> 4;
  int total = (K >> 5) * NT * 512;      // 512 halves per 32x16 tile
  for (int i = blockIdx.x * blockDim.x + threadIdx.x; i < total;
       i += gridDim.x * blockDim.x) {
    int tt = i >> 9, r = i & 511, l = r >> 4, e = r & 15;
    int kc = tt / NT, nt = tt % NT;
    int c = l & 15, kb = (l >> 4) << 3;
    int k = kc * 32 + kb + (e < 8 ? e : e + 8);
    out[i] = (_Float16)W[k * N + nt * 16 + c];
  }
}

// ---------------- main ACT kernel: one workgroup per sample ----------------
__global__ __launch_bounds__(256) void act_kernel(ActParams p) {
  const int t    = threadIdx.x;
  const int lane = t & 31;
  const int w    = t >> 5;
  const int s    = blockIdx.x;

  extern __shared__ char smem[];
  float*    S    = (float*)(smem + OFF_S);
  float*    SUM  = (float*)(smem + OFF_SUM);
  _Float16* U    = (_Float16*)(smem + OFF_U);
  _Float16* QH   = (_Float16*)(smem + OFF_QH);
  float*    SCR  = (float*)(smem + OFF_SCR);
  _Float16* P16  = (_Float16*)(smem + OFF_P);
  float*    MISC = (float*)(smem + OFF_MISC);
  float* gvec = MISC;          // 256
  float* hvec = MISC + 256;    // 32
  float* gmu  = MISC + 288;    // 32
  float* gvar = MISC + 320;    // 32
  float* scal = MISC + 352;    // [0]=halt_accum [1]=ponder [2]=step_prob

  // ---- init ----
  for (int i = t; i < 96 * SROW; i += 256) { S[i] = 0.f; SUM[i] = 0.f; }
  if (t == 0) { scal[0] = 0.f; scal[1] = 0.f; }

  // ---- constraint convs (scratch in SCR region) ----
  float* rowv = SCR;           // 48*9
  float* colv = SCR + 432;     // 48*9
  float* boxv = SCR + 864;     // 48*9 (3x3 per channel)
  float* xs   = SCR + 1296;    // 10*81
  __syncthreads();
  for (int i = t; i < 810; i += 256) xs[i] = p.x[s * 810 + i];
  __syncthreads();
  for (int i = t; i < 1296; i += 256) {
    int kind = i / 432, j = i % 432, co = j / 9, pos = j % 9;
    float acc;
    if (kind == 0) {                       // row constraint (1,9)
      acc = p.cc_br[co];
      for (int ci = 0; ci < 10; ci++)
        for (int kx = 0; kx < 9; kx++)
          acc += xs[ci * 81 + pos * 9 + kx] * p.cc_wr[(co * 10 + ci) * 9 + kx];
      rowv[co * 9 + pos] = fmaxf(acc, 0.f);
    } else if (kind == 1) {                // col constraint (9,1)
      acc = p.cc_bc[co];
      for (int ci = 0; ci < 10; ci++)
        for (int ky = 0; ky < 9; ky++)
          acc += xs[ci * 81 + ky * 9 + pos] * p.cc_wc[(co * 10 + ci) * 9 + ky];
      colv[co * 9 + pos] = fmaxf(acc, 0.f);
    } else {                               // box constraint 3x3 stride 3
      int br = pos / 3, bc = pos % 3;
      acc = p.cc_bb[co];
      for (int ci = 0; ci < 10; ci++)
        for (int ky = 0; ky < 3; ky++)
          for (int kx = 0; kx < 3; kx++)
            acc += xs[ci * 81 + (br * 3 + ky) * 9 + (bc * 3 + kx)] *
                   p.cc_wb[((co * 10 + ci) * 3 + ky) * 3 + kx];
      boxv[co * 9 + pos] = fmaxf(acc, 0.f);
    }
  }
  __syncthreads();
  // 1x1 reduce conv 144->128
  for (int i = t; i < 81 * 128; i += 256) {
    int cell = i >> 7, ch = i & 127;
    int r = cell / 9, cc = cell % 9, bi = (r / 3) * 3 + (cc / 3);
    const float* Wr = p.red_w + ch * 144;
    float acc = p.red_b[ch];
    for (int j = 0; j < 48; j++)
      acc += rowv[j * 9 + r] * Wr[j] + colv[j * 9 + cc] * Wr[48 + j] +
             boxv[j * 9 + bi] * Wr[96 + j];
    S[cell * SROW + ch] = acc;
  }
  __syncthreads();
  // group norm (32 groups of 4 channels, stats over 4*81)
  if (t < 32) {
    float m = 0.f, q = 0.f;
    for (int r = 0; r < 81; r++)
      for (int k = 0; k < 4; k++) {
        float v = S[r * SROW + t * 4 + k];
        m += v; q += v * v;
      }
    m /= 324.f;
    gmu[t] = m; gvar[t] = q / 324.f - m * m;
  }
  __syncthreads();
  for (int i = t; i < 81 * 128; i += 256) {
    int cell = i >> 7, ch = i & 127, g = ch >> 2;
    float v = (S[cell * SROW + ch] - gmu[g]) * rsqrtf(gvar[g] + 1e-5f) *
                  p.gn_s[ch] + p.gn_b[ch];
    S[cell * SROW + ch] = fmaxf(v, 0.f);
  }
  __syncthreads();

  // ================= ACT loop =================
  for (int step = 0; step < 16; step++) {
    // ---- halting probability (from current state) ----
    if (t < 128) {
      float m = 0.f, q = 0.f;
      for (int r = 0; r < 81; r++) { float v = S[r * SROW + t]; m += v; q += v * v; }
      float mu = m / 81.f;
      float var = (q - 81.f * mu * mu) / 80.f;          // ddof=1
      gvec[t] = mu; gvec[128 + t] = sqrtf(fmaxf(var, 0.f));
    }
    __syncthreads();
    if (t < 32) {
      float a = p.hg_b1[t];
      for (int i = 0; i < 256; i++) a += gvec[i] * p.hg_W1[i * 32 + t];
      hvec[t] = fmaxf(a, 0.f);
    }
    __syncthreads();
    if (t == 0) {
      float z = p.hg_b2[0];
      for (int i = 0; i < 32; i++) z += hvec[i] * p.hg_W2[i];
      float hp = 1.f / (1.f + __expf(-z));
      float sp = fminf(hp, 1.f - scal[0]);
      scal[2] = sp; scal[0] += sp; scal[1] += 1.f;
    }
    __syncthreads();
    float sp = scal[2];
    for (int i = t; i < 81 * 128; i += 256) {
      int cell = i >> 7, ch = i & 127;
      SUM[cell * SROW + ch] += S[cell * SROW + ch] * sp;   // state_sum += old_st * p
    }
    __syncthreads();

    // ---- LN1 -> U (f16) ----
    if (t < 96) {
      float m = 0.f, q = 0.f;
      for (int i = 0; i < 128; i++) { float v = S[t * SROW + i]; m += v; q += v * v; }
      m /= 128.f;
      float inv = rsqrtf(q / 128.f - m * m + 1e-5f);
      for (int i = 0; i < 128; i++) {
        float v = (S[t * SROW + i] - m) * inv * p.ln1_s[i] + p.ln1_b[i];
        U[t * 128 + i] = (_Float16)v;
      }
    }
    __syncthreads();

    // ---- QKV GEMM: [96x128] x [128x384] -> QH (f16) ----
    for (int tile = w; tile < 144; tile += WAVES) {
      int mt = tile / 24, nt = tile % 24, col = nt * 16 + (lane & 15);
      float bias = p.bqkv[col];
      v8f acc; for (int i = 0; i < 8; i++) acc[i] = bias;
      #pragma unroll
      for (int kc = 0; kc < 4; kc++) {
        v16h a = load_frag(U + mt * 16 * 128 + kc * 32, 128, lane);
        v16h b = *(const v16h*)(p.wqkv + ((kc * 24 + nt) * 32 + lane) * 16);
        acc = wmma16(a, b, acc);
      }
      int mo = (lane >> 4) << 3;
      for (int v = 0; v < 8; v++)
        QH[(mt * 16 + mo + v) * 384 + col] = (_Float16)acc[v];
    }
    __syncthreads();

    // ---- attention, per head ----
    for (int h = 0; h < 4; h++) {
      // scores = Q K^T / sqrt(dh)
      for (int tile = w; tile < 36; tile += WAVES) {
        int mt = tile / 6, nt = tile % 6;
        v8f acc; for (int i = 0; i < 8; i++) acc[i] = 0.f;
        v16h a = load_frag(QH + mt * 16 * 384 + h * 32, 384, lane);
        v16h b = load_frag(QH + nt * 16 * 384 + 128 + h * 32, 384, lane);
        acc = wmma16(a, b, acc);
        int mo = (lane >> 4) << 3, col = nt * 16 + (lane & 15);
        for (int v = 0; v < 8; v++)
          SCR[(mt * 16 + mo + v) * 96 + col] = acc[v] * 0.1767766953f;
      }
      __syncthreads();
      // row softmax over 81 real keys, pad keys masked out
      if (t < 96) {
        float mx = -1e30f;
        for (int c = 0; c < 81; c++) mx = fmaxf(mx, SCR[t * 96 + c]);
        float sm = 0.f;
        for (int c = 0; c < 81; c++) {
          float e = __expf(SCR[t * 96 + c] - mx);
          SCR[t * 96 + c] = e; sm += e;
        }
        float inv = 1.f / sm;
        for (int c = 0; c < 81; c++) P16[t * 96 + c] = (_Float16)(SCR[t * 96 + c] * inv);
        for (int c = 81; c < 96; c++) P16[t * 96 + c] = (_Float16)0.f;
      }
      __syncthreads();
      // O_head = P V  -> U (f16, per-head column slice)
      for (int tile = w; tile < 12; tile += WAVES) {
        int mt = tile / 2, nt = tile % 2;
        v8f acc; for (int i = 0; i < 8; i++) acc[i] = 0.f;
        #pragma unroll
        for (int kc = 0; kc < 3; kc++) {
          v16h a = load_frag(P16 + mt * 16 * 96 + kc * 32, 96, lane);
          v16h b = load_frag_kstrided(QH + kc * 32 * 384 + 256 + h * 32 + nt * 16,
                                      384, lane);
          acc = wmma16(a, b, acc);
        }
        int mo = (lane >> 4) << 3, col = h * 32 + nt * 16 + (lane & 15);
        for (int v = 0; v < 8; v++)
          U[(mt * 16 + mo + v) * 128 + col] = (_Float16)acc[v];
      }
      __syncthreads();
    }

    // ---- O-proj + residual: S += U @ Wo + bo ----
    for (int tile = w; tile < 48; tile += WAVES) {
      int mt = tile / 8, nt = tile % 8, col = nt * 16 + (lane & 15);
      float bias = p.bo[col];
      v8f acc; for (int i = 0; i < 8; i++) acc[i] = bias;
      #pragma unroll
      for (int kc = 0; kc < 4; kc++) {
        v16h a = load_frag(U + mt * 16 * 128 + kc * 32, 128, lane);
        v16h b = *(const v16h*)(p.wo + ((kc * 8 + nt) * 32 + lane) * 16);
        acc = wmma16(a, b, acc);
      }
      int mo = (lane >> 4) << 3;
      for (int v = 0; v < 8; v++) S[(mt * 16 + mo + v) * SROW + col] += acc[v];
    }
    __syncthreads();

    // ---- LN2 -> U (f16) ----
    if (t < 96) {
      float m = 0.f, q = 0.f;
      for (int i = 0; i < 128; i++) { float v = S[t * SROW + i]; m += v; q += v * v; }
      m /= 128.f;
      float inv = rsqrtf(q / 128.f - m * m + 1e-5f);
      for (int i = 0; i < 128; i++) {
        float v = (S[t * SROW + i] - m) * inv * p.ln2_s[i] + p.ln2_b[i];
        U[t * 128 + i] = (_Float16)v;
      }
    }
    __syncthreads();

    // ---- FFN1 + GELU -> H (QH region, stride 512) ----
    for (int tile = w; tile < 192; tile += WAVES) {
      int mt = tile / 32, nt = tile % 32, col = nt * 16 + (lane & 15);
      float bias = p.b1[col];
      v8f acc; for (int i = 0; i < 8; i++) acc[i] = bias;
      #pragma unroll
      for (int kc = 0; kc < 4; kc++) {
        v16h a = load_frag(U + mt * 16 * 128 + kc * 32, 128, lane);
        v16h b = *(const v16h*)(p.w1 + ((kc * 32 + nt) * 32 + lane) * 16);
        acc = wmma16(a, b, acc);
      }
      int mo = (lane >> 4) << 3;
      for (int v = 0; v < 8; v++) {
        float xv = acc[v];
        float g = 0.5f * xv *
                  (1.f + tanhf(0.7978845608f * (xv + 0.044715f * xv * xv * xv)));
        QH[(mt * 16 + mo + v) * 512 + col] = (_Float16)g;
      }
    }
    __syncthreads();

    // ---- FFN2 + residual: S += H @ W2 + b2 ----
    for (int tile = w; tile < 48; tile += WAVES) {
      int mt = tile / 8, nt = tile % 8, col = nt * 16 + (lane & 15);
      float bias = p.b2[col];
      v8f acc; for (int i = 0; i < 8; i++) acc[i] = bias;
      #pragma unroll
      for (int kc = 0; kc < 16; kc++) {
        v16h a = load_frag(QH + mt * 16 * 512 + kc * 32, 512, lane);
        v16h b = *(const v16h*)(p.w2 + ((kc * 8 + nt) * 32 + lane) * 16);
        acc = wmma16(a, b, acc);
      }
      int mo = (lane >> 4) << 3;
      for (int v = 0; v < 8; v++) S[(mt * 16 + mo + v) * SROW + col] += acc[v];
    }
    __syncthreads();

    if (scal[0] >= HALT_T) break;   // uniform: halted sample stops exactly per reference
  }

  // ---- remainder + final FC ----
  __syncthreads();
  float rem = 1.f - scal[0];
  for (int i = t; i < 81 * 128; i += 256) {
    int cell = i >> 7, ch = i & 127;
    SUM[cell * SROW + ch] += S[cell * SROW + ch] * rem;
  }
  if (t == 0) scal[1] += rem;
  __syncthreads();
  for (int i = t; i < 729; i += 256) {
    int cell = i / 9, j = i % 9;
    float acc = p.fc_b[j];
    for (int d = 0; d < 128; d++) acc += SUM[cell * SROW + d] * p.fc_W[d * 9 + j];
    p.out[s * 729 + i] = acc;
  }
  if (t == 0) p.out[p.nb * 729 + s] = scal[1];
}

// ---------------- host launch ----------------
extern "C" void kernel_launch(void* const* d_in, const int* in_sizes, int n_in,
                              void* d_out, int out_size, void* d_ws, size_t ws_size,
                              hipStream_t stream) {
  (void)n_in; (void)out_size; (void)ws_size;
  const int Bn = in_sizes[0] / 810;

  _Float16* ws16 = (_Float16*)d_ws;
  // d_in order: x, cc_wr, cc_br, cc_wc, cc_bc, cc_wb, cc_bb, red_w, red_b,
  // gn_s, gn_b, ln1_s, ln1_b, Wqkv, bqkv, Wo, bo, ln2_s, ln2_b, W1, b1, W2, b2,
  // hg_W1, hg_b1, hg_W2, hg_b2, fc_W, fc_b
  const float* Wqkv = (const float*)d_in[13];
  const float* Wo   = (const float*)d_in[15];
  const float* W1   = (const float*)d_in[19];
  const float* W2   = (const float*)d_in[21];

  // pre-swizzle weights into WMMA B-fragment layout (f16) in workspace
  swz_kernel<<<96, 256, 0, stream>>>(Wqkv, ws16 + 0,      128, 384);
  swz_kernel<<<32, 256, 0, stream>>>(Wo,   ws16 + 49152,  128, 128);
  swz_kernel<<<128, 256, 0, stream>>>(W1,  ws16 + 65536,  128, 512);
  swz_kernel<<<128, 256, 0, stream>>>(W2,  ws16 + 131072, 512, 128);

  ActParams p;
  p.x     = (const float*)d_in[0];
  p.cc_wr = (const float*)d_in[1];  p.cc_br = (const float*)d_in[2];
  p.cc_wc = (const float*)d_in[3];  p.cc_bc = (const float*)d_in[4];
  p.cc_wb = (const float*)d_in[5];  p.cc_bb = (const float*)d_in[6];
  p.red_w = (const float*)d_in[7];  p.red_b = (const float*)d_in[8];
  p.gn_s  = (const float*)d_in[9];  p.gn_b  = (const float*)d_in[10];
  p.ln1_s = (const float*)d_in[11]; p.ln1_b = (const float*)d_in[12];
  p.bqkv  = (const float*)d_in[14];
  p.bo    = (const float*)d_in[16];
  p.ln2_s = (const float*)d_in[17]; p.ln2_b = (const float*)d_in[18];
  p.b1    = (const float*)d_in[20]; p.b2    = (const float*)d_in[22];
  p.hg_W1 = (const float*)d_in[23]; p.hg_b1 = (const float*)d_in[24];
  p.hg_W2 = (const float*)d_in[25]; p.hg_b2 = (const float*)d_in[26];
  p.fc_W  = (const float*)d_in[27]; p.fc_b  = (const float*)d_in[28];
  p.wqkv = ws16;          p.wo = ws16 + 49152;
  p.w1   = ws16 + 65536;  p.w2 = ws16 + 131072;
  p.out  = (float*)d_out;
  p.nb   = Bn;

  act_kernel<<<Bn, 256, SMEM_BYTES, stream>>>(p);
}
